// RelationClassifier_12970801234000
// MI455X (gfx1250) — matrix operand order repaired and checked by reading, compiled, not run on hardware
//
#include <hip/hip_runtime.h>
#include <hip/hip_bf16.h>

// ---------------------------------------------------------------------------
// RelationClassifier on gfx1250 (MI455X). bf16 WMMA GEMMs, fp32 accumulate.
// All staging buffers padded so the GEMM has zero bounds checks: fragments are
// unconditional b128 loads (A row-major MxK, B transposed NxK, both
// K-contiguous per lane).
// ---------------------------------------------------------------------------

typedef __attribute__((ext_vector_type(16))) unsigned short v16u;
typedef __attribute__((ext_vector_type(16))) __bf16         v16bf;
typedef __attribute__((ext_vector_type(8)))  float          v8f;

union Frag { v16bf b; v16u u; uint4 q[2]; };

__device__ __forceinline__ unsigned short f2bf(float f) {
  unsigned u = __float_as_uint(f);
  unsigned r = (u + 0x7FFFu + ((u >> 16) & 1u)) >> 16;   // round-to-nearest-even
  return (unsigned short)r;
}

// ---------------------------------------------------------------------------
// GEMM: C[z] = scale * (A[z] @ Bt[z]^T) + bias
//   A : M x K bf16, row-major, stride lda   (M % 64 == 0, K % 32 == 0)
//   Bt: N x K bf16, row-major, stride ldbt  (N % 16 == 0)
// One wave: 64(M) x 16(N) via 4x v_wmma_f32_16x16x32_bf16 per K-step,
// sharing the B fragment. EXEC stays all-ones (no divergence).
// ---------------------------------------------------------------------------
__global__ void gemm_bf16_wmma(
    const unsigned short* __restrict__ A, const unsigned short* __restrict__ Bt,
    float* __restrict__ C, unsigned short* __restrict__ Cbf,
    const float* __restrict__ bias,
    int N, int K, int lda, int ldbt, int ldc, int ldcbf, float scale,
    long long aS, long long bS, long long cS, long long cbS)
{
  const int z = blockIdx.z;
  const unsigned short* Ab = A  + z * aS;
  const unsigned short* Bb = Bt + z * bS;

  const int lane = threadIdx.x;                      // 0..31
  const int n0 = (blockIdx.x * 4 + threadIdx.y) * 16;
  if (n0 >= N) return;                               // wave-uniform
  const int m0 = blockIdx.y * 64;
  const int hl  = lane >> 4;                         // half-wave 0/1
  const int l15 = lane & 15;

  // Per-lane row pointers (ISA 7.12.2 fragment layouts).
  const unsigned short* ar0 = Ab + (long long)(m0 + l15) * lda;
  const unsigned short* ar1 = ar0 + (long long)16 * lda;
  const unsigned short* ar2 = ar0 + (long long)32 * lda;
  const unsigned short* ar3 = ar0 + (long long)48 * lda;
  const unsigned short* br  = Bb + (long long)(n0 + l15) * ldbt;

  v8f acc[4];
  acc[0] = {}; acc[1] = {}; acc[2] = {}; acc[3] = {};

  for (int k0 = 0; k0 < K; k0 += 32) {
    // B fragment: lane n=col, elements K = k0 + {0..7|16..23} + 8*hl
    Frag b;
    b.q[0] = *(const uint4*)(br + k0 + 8 * hl);
    b.q[1] = *(const uint4*)(br + k0 + 16 + 8 * hl);
    // A fragments: lane m=row, elements K = k0 + {0..7|16..23} + 8*hl
    const int ka0 = k0 + 8 * hl, ka1 = k0 + 16 + 8 * hl;
    Frag a0, a1, a2, a3;
    a0.q[0] = *(const uint4*)(ar0 + ka0); a0.q[1] = *(const uint4*)(ar0 + ka1);
    a1.q[0] = *(const uint4*)(ar1 + ka0); a1.q[1] = *(const uint4*)(ar1 + ka1);
    a2.q[0] = *(const uint4*)(ar2 + ka0); a2.q[1] = *(const uint4*)(ar2 + ka1);
    a3.q[0] = *(const uint4*)(ar3 + ka0); a3.q[1] = *(const uint4*)(ar3 + ka1);

    acc[0] = __builtin_amdgcn_wmma_f32_16x16x32_bf16(false, a0.b, false, b.b, (short)0, acc[0], false, false);
    acc[1] = __builtin_amdgcn_wmma_f32_16x16x32_bf16(false, a1.b, false, b.b, (short)0, acc[1], false, false);
    acc[2] = __builtin_amdgcn_wmma_f32_16x16x32_bf16(false, a2.b, false, b.b, (short)0, acc[2], false, false);
    acc[3] = __builtin_amdgcn_wmma_f32_16x16x32_bf16(false, a3.b, false, b.b, (short)0, acc[3], false, false);
  }

  // C/D layout: VGPR r -> row m0 + 16*t + r + 8*hl, col n0 + (lane&15)
  const int n = n0 + l15;
  const float bv = bias ? bias[n] : 0.f;
#pragma unroll
  for (int t = 0; t < 4; ++t) {
#pragma unroll
    for (int r = 0; r < 8; ++r) {
      const int m = m0 + t * 16 + r + 8 * hl;
      const float v = acc[t][r] * scale + bv;
      if (C)   C  [cS  * z + (long long)m * ldc   + n] = v;
      if (Cbf) Cbf[cbS * z + (long long)m * ldcbf + n] = f2bf(v);
    }
  }
}

// ---------------------------------------------------------------------------
// staging kernels
// ---------------------------------------------------------------------------
__global__ void embed_concat_kernel(const float* __restrict__ x,
                                    const int* __restrict__ labels,
                                    const float* __restrict__ ent,
                                    unsigned short* __restrict__ outbf,
                                    int BS, int IN, int EMB)
{
  const int Dd = IN + EMB;
  long long i = (long long)blockIdx.x * blockDim.x + threadIdx.x;
  if (i >= (long long)BS * Dd) return;
  int row = (int)(i / Dd), c = (int)(i % Dd);
  float v = (c < IN) ? x[(long long)row * IN + c]
                     : ent[labels[row] * EMB + (c - IN)];
  outbf[i] = f2bf(v);
}

// out[n*Kpad + k] = bf16(in[k*Ncols + n]) for k < Kdim else 0   (B transpose+pad)
__global__ void castT_kernel(const float* __restrict__ in,
                             unsigned short* __restrict__ out,
                             int Kdim, int Ncols, int Kpad)
{
  long long i = (long long)blockIdx.x * blockDim.x + threadIdx.x;
  if (i >= (long long)Ncols * Kpad) return;
  int k = (int)(i % Kpad), n = (int)(i / Kpad);
  out[i] = (k < Kdim) ? f2bf(in[(long long)k * Ncols + n]) : (unsigned short)0;
}

// out[r*Cpad + c] = bf16(in[r*Cc + c]) for r<R && c<Cc else 0   (A pad)
__global__ void cast_pad_kernel(const float* __restrict__ in,
                                unsigned short* __restrict__ out,
                                int R, int Cc, int Cpad, long long tot)
{
  long long i = (long long)blockIdx.x * blockDim.x + threadIdx.x;
  if (i >= tot) return;
  int c = (int)(i % Cpad); long long r = i / Cpad;
  out[i] = (r < R && c < Cc) ? f2bf(in[r * Cc + c]) : (unsigned short)0;
}

// qkv (1024 x 2400, head-interleaved) -> q_pad/k_pad (z,512,256) and vT_pad (z,256,512)
__global__ void qkv_repack_kernel(const unsigned short* __restrict__ qkv,
                                  unsigned short* __restrict__ q_pad,
                                  unsigned short* __restrict__ k_pad,
                                  unsigned short* __restrict__ vT_pad)
{
  long long i = (long long)blockIdx.x * blockDim.x + threadIdx.x;
  if (i >= 8LL * 512 * 256) return;
  const int d = (int)(i & 255);
  const int s = (int)((i >> 8) & 511);
  const int z = (int)(i >> 17);
  const int b = z >> 2, h = z & 3;
  unsigned short qv = 0, kv = 0, vv = 0;
  if (d < 200) {
    const long long base = ((long long)(b * 512 + s)) * 2400 + h * 200 + d;
    qv = qkv[base]; kv = qkv[base + 800]; vv = qkv[base + 1600];
  }
  q_pad[i] = qv;
  k_pad[i] = kv;
  vT_pad[((long long)z * 256 + d) * 512 + s] = vv;
}

// o_pad (z,512,256) -> o_bf (1024 x 800)
__global__ void o_repack_kernel(const unsigned short* __restrict__ o_pad,
                                unsigned short* __restrict__ o_bf)
{
  long long i = (long long)blockIdx.x * blockDim.x + threadIdx.x;
  if (i >= 1024LL * 800) return;
  const int c = (int)(i % 800);
  const int row = (int)(i / 800);
  const int b = row >> 9, s = row & 511;
  const int h = c / 200, d = c % 200;
  o_bf[i] = o_pad[(((long long)(b * 4 + h)) * 512 + s) * 256 + d];
}

// ---------------------------------------------------------------------------
// row-wise LayerNorm (optional pre-ReLU) -> bf16
// ---------------------------------------------------------------------------
__global__ void ln_kernel(const float* __restrict__ in,
                          const float* __restrict__ g, const float* __restrict__ be,
                          unsigned short* __restrict__ outbf,
                          int width, int do_relu)
{
  __shared__ float red[256];
  const int row = blockIdx.x;
  const float* r = in + (long long)row * width;
  float s = 0.f, s2 = 0.f;
  for (int c = threadIdx.x; c < width; c += blockDim.x) {
    float v = r[c]; if (do_relu) v = fmaxf(v, 0.f);
    s += v; s2 += v * v;
  }
  red[threadIdx.x] = s; __syncthreads();
  for (int st = blockDim.x / 2; st > 0; st >>= 1) {
    if (threadIdx.x < st) red[threadIdx.x] += red[threadIdx.x + st];
    __syncthreads();
  }
  const float mean = red[0] / width;
  __syncthreads();
  red[threadIdx.x] = s2; __syncthreads();
  for (int st = blockDim.x / 2; st > 0; st >>= 1) {
    if (threadIdx.x < st) red[threadIdx.x] += red[threadIdx.x + st];
    __syncthreads();
  }
  const float var  = red[0] / width - mean * mean;
  const float rstd = rsqrtf(var + 1e-5f);
  for (int c = threadIdx.x; c < width; c += blockDim.x) {
    float v = r[c]; if (do_relu) v = fmaxf(v, 0.f);
    outbf[(long long)row * width + c] = f2bf((v - mean) * rstd * g[c] + be[c]);
  }
}

// ---------------------------------------------------------------------------
// softmax over keys: attn = softmax(scores + sentence_mask) -> bf16
// ---------------------------------------------------------------------------
__global__ void softmax_kernel(const float* __restrict__ scores,
                               const float* __restrict__ smask,
                               unsigned short* __restrict__ attnbf,
                               int S, int H)
{
  __shared__ float red[128];
  __shared__ float vals[512];
  const int z = blockIdx.y, q = blockIdx.x;
  const int b = z / H;
  const float* sr = scores + ((long long)z * S + q) * S;
  const float* mr = smask  + ((long long)b * S + q) * S;
  const int tid = threadIdx.x, bd = blockDim.x;

  float mx = -3.0e38f;
  for (int c = tid; c < S; c += bd) {
    float v = sr[c] + mr[c];
    vals[c] = v;
    mx = fmaxf(mx, v);
  }
  red[tid] = mx; __syncthreads();
  for (int st = bd / 2; st > 0; st >>= 1) {
    if (tid < st) red[tid] = fmaxf(red[tid], red[tid + st]);
    __syncthreads();
  }
  mx = red[0]; __syncthreads();

  float sum = 0.f;
  for (int c = tid; c < S; c += bd) {
    float e = __expf(vals[c] - mx);
    vals[c] = e;
    sum += e;
  }
  red[tid] = sum; __syncthreads();
  for (int st = bd / 2; st > 0; st >>= 1) {
    if (tid < st) red[tid] += red[tid + st];
    __syncthreads();
  }
  const float inv = 1.f / red[0];
  __syncthreads();
  for (int c = tid; c < S; c += bd)
    attnbf[((long long)z * S + q) * S + c] = f2bf(vals[c] * inv);
}

// ---------------------------------------------------------------------------
// pair stage: logits[b,i,j,:] = mask * (relu(hi[b,i]+hj[b,j]+hp[i-j+511]) @ w2 + b2)
// ---------------------------------------------------------------------------
__global__ void pair_logits_kernel(const float* __restrict__ hi,
                                   const float* __restrict__ hj,
                                   const float* __restrict__ hp,
                                   const float* __restrict__ w2,
                                   const float* __restrict__ b2,
                                   const unsigned char* __restrict__ amask,
                                   float* __restrict__ out, int S)
{
  __shared__ float hi_s[256];
  __shared__ float w2_s[256 * 5];
  __shared__ float b2_s[5];
  const int b = blockIdx.z, i = blockIdx.y;
  const int tid = threadIdx.x;              // 0..63
  for (int c = tid; c < 256; c += 64)  hi_s[c] = hi[((long long)b * S + i) * 256 + c];
  for (int c = tid; c < 1280; c += 64) w2_s[c] = w2[c];
  if (tid < 5) b2_s[tid] = b2[tid];
  __syncthreads();

  const int j = blockIdx.x * 64 + tid;
  const int rel = i - j + 511;              // MAX_SEQ_LEN-1
  const float4* hj4 = (const float4*)(hj + ((long long)b * S + j) * 256);
  const float4* hp4 = (const float4*)(hp + (long long)rel * 256);

  float a0 = b2_s[0], a1 = b2_s[1], a2 = b2_s[2], a3 = b2_s[3], a4 = b2_s[4];
  for (int c4 = 0; c4 < 64; ++c4) {
    const float4 vj = hj4[c4];
    const float4 vp = hp4[c4];
    const int c = c4 * 4;
    float h;
    h = fmaxf(hi_s[c + 0] + vj.x + vp.x, 0.f);
    a0 += h * w2_s[(c + 0) * 5 + 0]; a1 += h * w2_s[(c + 0) * 5 + 1]; a2 += h * w2_s[(c + 0) * 5 + 2];
    a3 += h * w2_s[(c + 0) * 5 + 3]; a4 += h * w2_s[(c + 0) * 5 + 4];
    h = fmaxf(hi_s[c + 1] + vj.y + vp.y, 0.f);
    a0 += h * w2_s[(c + 1) * 5 + 0]; a1 += h * w2_s[(c + 1) * 5 + 1]; a2 += h * w2_s[(c + 1) * 5 + 2];
    a3 += h * w2_s[(c + 1) * 5 + 3]; a4 += h * w2_s[(c + 1) * 5 + 4];
    h = fmaxf(hi_s[c + 2] + vj.z + vp.z, 0.f);
    a0 += h * w2_s[(c + 2) * 5 + 0]; a1 += h * w2_s[(c + 2) * 5 + 1]; a2 += h * w2_s[(c + 2) * 5 + 2];
    a3 += h * w2_s[(c + 2) * 5 + 3]; a4 += h * w2_s[(c + 2) * 5 + 4];
    h = fmaxf(hi_s[c + 3] + vj.w + vp.w, 0.f);
    a0 += h * w2_s[(c + 3) * 5 + 0]; a1 += h * w2_s[(c + 3) * 5 + 1]; a2 += h * w2_s[(c + 3) * 5 + 2];
    a3 += h * w2_s[(c + 3) * 5 + 3]; a4 += h * w2_s[(c + 3) * 5 + 4];
  }
  const float mv = amask[((long long)b * S + i) * S + j] ? 1.f : 0.f;
  long long o = (((long long)b * S + i) * S + j) * 5;
  out[o + 0] = a0 * mv; out[o + 1] = a1 * mv; out[o + 2] = a2 * mv;
  out[o + 3] = a3 * mv; out[o + 4] = a4 * mv;
}

// ---------------------------------------------------------------------------
// host-side orchestration
// ---------------------------------------------------------------------------
extern "C" void kernel_launch(void* const* d_in, const int* in_sizes, int n_in,
                              void* d_out, int out_size, void* d_ws, size_t ws_size,
                              hipStream_t stream) {
  (void)in_sizes; (void)n_in; (void)out_size; (void)ws_size;

  const float* x      = (const float*)d_in[0];
  const int*   labels = (const int*)  d_in[1];
  const unsigned char* amask = (const unsigned char*)d_in[2];  // jnp bool -> 1 byte
  const float* smask  = (const float*)d_in[3];
  const float* ent    = (const float*)d_in[4];
  const float* w_t    = (const float*)d_in[5];
  const float* b_t    = (const float*)d_in[6];
  const float* g1     = (const float*)d_in[7];
  const float* be1    = (const float*)d_in[8];
  const float* w_in   = (const float*)d_in[9];
  const float* b_in   = (const float*)d_in[10];
  const float* w_out  = (const float*)d_in[11];
  const float* b_out  = (const float*)d_in[12];
  const float* g2     = (const float*)d_in[13];
  const float* be2    = (const float*)d_in[14];
  const float* w1     = (const float*)d_in[15];
  const float* b1     = (const float*)d_in[16];
  const float* w2     = (const float*)d_in[17];
  const float* b2     = (const float*)d_in[18];
  const float* rel_t  = (const float*)d_in[19];
  float* out = (float*)d_out;

  const int Bsz = 2, S = 512, IN = 768, EMB = 32, D = 800, H = 4;
  const int BS = Bsz * S, D3 = 3 * D, HID = 256, REL = 2 * 512 - 1;
  const int Z = Bsz * H;                     // 8 attention batches
  const int HDP = 256;                       // head_dim 200 padded to 256

  // workspace arena (256B aligned carves; all bf16 rows stay 16B aligned)
  size_t off = 0;
  char* basep = (char*)d_ws;
  auto carve = [&](size_t bytes) -> void* {
    off = (off + 255) & ~(size_t)255;
    void* r = basep + off;
    off += bytes;
    return r;
  };
  unsigned short* combined_bf = (unsigned short*)carve((size_t)BS * D * 2);
  unsigned short* wtT   = (unsigned short*)carve((size_t)D * D * 2);
  unsigned short* winT  = (unsigned short*)carve((size_t)D3 * D * 2);
  unsigned short* woutT = (unsigned short*)carve((size_t)D * D * 2);
  unsigned short* w1aT  = (unsigned short*)carve((size_t)HID * D * 2);
  unsigned short* w1bT  = (unsigned short*)carve((size_t)HID * D * 2);
  unsigned short* w1cT  = (unsigned short*)carve((size_t)HID * 32 * 2);
  unsigned short* rel_pad = (unsigned short*)carve((size_t)1024 * 32 * 2);
  float*          c1    = (float*)         carve((size_t)BS * D * 4);
  unsigned short* t_bf  = (unsigned short*)carve((size_t)BS * D * 2);
  unsigned short* qkv_bf= (unsigned short*)carve((size_t)BS * D3 * 2);
  unsigned short* q_pad = (unsigned short*)carve((size_t)Z * S * HDP * 2);
  unsigned short* k_pad = (unsigned short*)carve((size_t)Z * S * HDP * 2);
  unsigned short* vT_pad= (unsigned short*)carve((size_t)Z * HDP * S * 2);
  float*          scores= (float*)         carve((size_t)Z * S * S * 4);
  unsigned short* attn_bf=(unsigned short*)carve((size_t)Z * S * S * 2);
  unsigned short* o_pad = (unsigned short*)carve((size_t)Z * S * HDP * 2);
  unsigned short* o_bf  = (unsigned short*)carve((size_t)BS * D * 2);
  float*          o2    = (float*)         carve((size_t)BS * D * 4);
  unsigned short* o2_bf = (unsigned short*)carve((size_t)BS * D * 2);
  float*          hi    = (float*)         carve((size_t)BS * HID * 4);
  float*          hj    = (float*)         carve((size_t)BS * HID * 4);
  float*          hp    = (float*)         carve((size_t)1024 * HID * 4);

  auto gemm = [&](const unsigned short* A, const unsigned short* Bt,
                  float* C, unsigned short* Cbf, const float* bias,
                  int M, int N, int K, int lda, int ldbt, int ldc, int ldcbf,
                  float scale, int batches,
                  long long aS, long long bS, long long cS, long long cbS) {
    dim3 grid((N + 63) / 64, M / 64, batches);
    dim3 block(32, 4, 1);
    gemm_bf16_wmma<<<grid, block, 0, stream>>>(A, Bt, C, Cbf, bias,
        N, K, lda, ldbt, ldc, ldcbf, scale, aS, bS, cS, cbS);
  };
  auto castT = [&](const float* in, unsigned short* dst, int Kdim, int Ncols, int Kpad) {
    long long n = (long long)Ncols * Kpad;
    castT_kernel<<<(int)((n + 255) / 256), 256, 0, stream>>>(in, dst, Kdim, Ncols, Kpad);
  };

  // 0) weights -> bf16, transposed (N x K) with K padding
  castT(w_t,   wtT,   D,  D,  D);
  castT(w_in,  winT,  D,  D3, D);
  castT(w_out, woutT, D,  D,  D);
  castT(w1,                    w1aT, D,  HID, D);
  castT(w1 + (size_t)D  * HID, w1bT, D,  HID, D);
  castT(w1 + (size_t)2*D* HID, w1cT, 20, HID, 32);
  cast_pad_kernel<<<(1024 * 32 + 255) / 256, 256, 0, stream>>>(
      rel_t, rel_pad, REL, 20, 32, 1024LL * 32);

  // 1) combined = concat(x, ent_emb[labels]) -> bf16 (1024 x 800)
  {
    long long n = (long long)BS * D;
    embed_concat_kernel<<<(int)((n + 255) / 256), 256, 0, stream>>>(
        x, labels, ent, combined_bf, BS, IN, EMB);
  }

  // 2) c1 = combined @ w_t + b_t   (f32)
  gemm(combined_bf, wtT, c1, nullptr, b_t, BS, D, D, D, D, D, 0, 1.f, 1, 0,0,0,0);
  // 3) t = LN(relu(c1)) -> bf16
  ln_kernel<<<BS, 256, 0, stream>>>(c1, g1, be1, t_bf, D, 1);

  // 4) qkv = t @ w_in + b_in -> bf16 (1024 x 2400)
  gemm(t_bf, winT, nullptr, qkv_bf, b_in, BS, D3, D, D, D, 0, D3, 1.f, 1, 0,0,0,0);
  // 4b) repack to padded per-head layouts
  qkv_repack_kernel<<<(int)((8LL * S * HDP + 255) / 256), 256, 0, stream>>>(
      qkv_bf, q_pad, k_pad, vT_pad);

  // 5) scores[z] = q @ k^T / sqrt(200)
  gemm(q_pad, k_pad, scores, nullptr, nullptr, S, S, HDP, HDP, HDP, S, 0,
       0.070710678f, Z,
       (long long)S * HDP, (long long)S * HDP, (long long)S * S, 0);

  // 6) attn = softmax(scores + sentence_mask) -> bf16
  {
    dim3 grid(S, Z);
    softmax_kernel<<<grid, 128, 0, stream>>>(scores, smask, attn_bf, S, H);
  }

  // 7) o_pad[z] = attn[z] @ v[z]   (N padded to 256; pad cols are zero)
  gemm(attn_bf, vT_pad, nullptr, o_pad, nullptr, S, HDP, S, S, S, 0, HDP,
       1.f, Z,
       (long long)S * S, (long long)HDP * S, 0, (long long)S * HDP);
  o_repack_kernel<<<(int)((1024LL * 800 + 255) / 256), 256, 0, stream>>>(o_pad, o_bf);

  // 8) o2 = o @ w_out + b_out; LN -> bf16
  gemm(o_bf, woutT, o2, nullptr, b_out, BS, D, D, D, D, D, 0, 1.f, 1, 0,0,0,0);
  ln_kernel<<<BS, 256, 0, stream>>>(o2, g2, be2, o2_bf, D, 0);

  // 9) hi = o2 @ w1[:D] + b1 ; hj = o2 @ w1[D:2D] ; hp = rel_pad @ w1[2D:]
  gemm(o2_bf,   w1aT, hi, nullptr, b1,      BS,   HID, D,  D,  D,  HID, 0, 1.f, 1, 0,0,0,0);
  gemm(o2_bf,   w1bT, hj, nullptr, nullptr, BS,   HID, D,  D,  D,  HID, 0, 1.f, 1, 0,0,0,0);
  gemm(rel_pad, w1cT, hp, nullptr, nullptr, 1024, HID, 32, 32, 32, HID, 0, 1.f, 1, 0,0,0,0);

  // 10) logits = mask * (relu(hi+hj+hp) @ w2 + b2)
  {
    dim3 grid(S / 64, S, Bsz);
    pair_logits_kernel<<<grid, 64, 0, stream>>>(hi, hj, hp, w2, b2, amask, out, S);
  }
}
